// TimeAwareSelfAttention_36507222016344
// MI455X (gfx1250) — matrix operand order repaired and checked by reading, not compiled, not run on hardware
//
#include <hip/hip_runtime.h>
#include <hip/hip_bf16.h>

typedef __attribute__((ext_vector_type(16))) __bf16 v16bf;
typedef __attribute__((ext_vector_type(8)))  __bf16 v8bf;
typedef __attribute__((ext_vector_type(8)))  float   v8f;

#define BATCH  16
#define CCH    512
#define SEQ    1024          // H*W = 32*32
#define NHEAD  8
#define HDIM   64
#define INNER  512           // NHEAD*HDIM
#define ROWS   (BATCH*SEQ)   // 16384

static __device__ __forceinline__ v8f wmma_bf16(v16bf a, v16bf b, v8f c) {
  // (neg_a, A, neg_b, B, c_mod, C, reuse_a, reuse_b)
  return __builtin_amdgcn_wmma_f32_16x16x32_bf16(false, a, false, b, (short)0, c, false, false);
}
static __device__ __forceinline__ v8bf ldg8(const __bf16* p) {
  return *reinterpret_cast<const v8bf*>(p);
}
static __device__ __forceinline__ v16bf cat8(v8bf lo, v8bf hi) {
  return __builtin_shufflevector(lo, hi, 0,1,2,3,4,5,6,7,8,9,10,11,12,13,14,15);
}
static __device__ __forceinline__ float redmax16(float v) {
  v = fmaxf(v, __shfl_xor(v, 1, 16));
  v = fmaxf(v, __shfl_xor(v, 2, 16));
  v = fmaxf(v, __shfl_xor(v, 4, 16));
  v = fmaxf(v, __shfl_xor(v, 8, 16));
  return v;
}
static __device__ __forceinline__ float redsum16(float v) {
  v += __shfl_xor(v, 1, 16);
  v += __shfl_xor(v, 2, 16);
  v += __shfl_xor(v, 4, 16);
  v += __shfl_xor(v, 8, 16);
  return v;
}

// ---------------- tiny pointwise / MLP kernels ----------------

// hidden = silu(time_emb @ w1 + b1)   (16 x 512)
__global__ void k_mlp1(const float* __restrict__ te, const float* __restrict__ w1,
                       const float* __restrict__ b1, float* __restrict__ hid) {
  int idx = blockIdx.x * blockDim.x + threadIdx.x;   // 16*512
  int b = idx >> 9, j = idx & 511;
  float acc = b1[j];
  const float* t = te + b * 512;
  for (int i = 0; i < 512; ++i) acc += t[i] * w1[i * 512 + j];
  hid[idx] = acc / (1.0f + __expf(-acc));            // silu
}

// gb = hidden @ w2 + b2   (16 x 1024) -> gamma|beta
__global__ void k_mlp2(const float* __restrict__ hid, const float* __restrict__ w2,
                       const float* __restrict__ b2, float* __restrict__ gb) {
  int idx = blockIdx.x * blockDim.x + threadIdx.x;   // 16*1024
  int b = idx >> 10, j = idx & 1023;
  float acc = b2[j];
  const float* h = hid + b * 512;
  for (int i = 0; i < 512; ++i) acc += h[i] * w2[i * 1024 + j];
  gb[idx] = acc;
}

// per-pixel channel mean / rstd over C=512 (coalesced across pixels)
__global__ void k_stats(const float* __restrict__ x, float* __restrict__ stats) {
  int blk = blockIdx.x;                 // 64 blocks of 256 threads
  int b = blk >> 2;
  int n = ((blk & 3) << 8) + threadIdx.x;
  float s = 0.f, ss = 0.f;
  for (int c = 0; c < 512; ++c) {
    float v = x[(size_t)(b * 512 + c) * 1024 + n];
    s += v; ss += v * v;
  }
  float mean = s * (1.f / 512.f);
  float var  = ss * (1.f / 512.f) - mean * mean;
  int row = b * 1024 + n;
  stats[row * 2 + 0] = mean;
  stats[row * 2 + 1] = rsqrtf(var + 1e-6f);
}

// AdaCLN -> tokens (ROWS x C) bf16, row-major
__global__ void k_tokens(const float* __restrict__ x, const float* __restrict__ stats,
                         const float* __restrict__ gb, __bf16* __restrict__ tok) {
  int row = blockIdx.x;                 // 16384
  int b = row >> 10, n = row & 1023;
  float mean = stats[row * 2], rstd = stats[row * 2 + 1];
  for (int c = threadIdx.x; c < 512; c += blockDim.x) {
    float v  = (x[(size_t)(b * 512 + c) * 1024 + n] - mean) * rstd;
    float g  = gb[b * 1024 + c];
    float be = gb[b * 1024 + 512 + c];
    tok[(size_t)row * 512 + c] = (__bf16)(v * (1.f + g) + be);
  }
}

// transpose + fp32->bf16: out[c*R + r] = in[r*C + c]   (pack W^T for B-fragments)
__global__ void k_packT(const float* __restrict__ in, __bf16* __restrict__ out,
                        int R, int C) {
  int idx = blockIdx.x * blockDim.x + threadIdx.x;
  if (idx >= R * C) return;
  int r = idx / C, c = idx % C;
  out[(size_t)c * R + r] = (__bf16)in[idx];
}

// 2-D RoPE cos/sin table (SEQ x HDIM)
__global__ void k_ropetab(float* __restrict__ cosT, float* __restrict__ sinT) {
  int idx = blockIdx.x * blockDim.x + threadIdx.x;   // 1024*64
  int n = idx >> 6, t = idx & 63;
  int y = n >> 5, xp = n & 31;
  float pos = (t < 32) ? (float)y : (float)xp;
  int j = t & 15;
  float inv = __powf(100.0f, -(float)j * (1.0f / 16.0f));
  float ang = pos * inv;
  cosT[idx] = __cosf(ang);
  sinT[idx] = __sinf(ang);
}

// in-place RoPE on q,k (per-head rows of 64); each thread handles pair (t, t+32)
__global__ void k_rope(__bf16* __restrict__ q, __bf16* __restrict__ k,
                       const float* __restrict__ cosT, const float* __restrict__ sinT) {
  int idx = blockIdx.x * blockDim.x + threadIdx.x;   // B*8*1024*32
  int t = idx & 31;
  int n = (idx >> 5) & 1023;
  size_t base = (size_t)(idx >> 5) * 64;
  float c0 = cosT[n * 64 + t],      s0 = sinT[n * 64 + t];
  float c1 = cosT[n * 64 + t + 32], s1 = sinT[n * 64 + t + 32];
  float q0 = (float)q[base + t], q1 = (float)q[base + t + 32];
  q[base + t]      = (__bf16)(q0 * c0 - q1 * s0);
  q[base + t + 32] = (__bf16)(q1 * c1 + q0 * s1);
  float k0 = (float)k[base + t], k1 = (float)k[base + t + 32];
  k[base + t]      = (__bf16)(k0 * c0 - k1 * s0);
  k[base + t + 32] = (__bf16)(k1 * c1 + k0 * s1);
}

// ---------------- WMMA GEMM kernels ----------------

// qkv = tokens(16384x512) @ qkv_w(512x1536); Bt is packed W^T (1536x512).
// Wave tile: 16 rows x 64 cols (A-frag reused over 4 B-frags per k-step).
// Epilogue scatters to q[b,h,n,d], k[b,h,n,d], vt[b,h,d,n] (bf16).
__global__ void __launch_bounds__(256) k_qkv_gemm(const __bf16* __restrict__ A,
                                                  const __bf16* __restrict__ Bt,
                                                  __bf16* __restrict__ q,
                                                  __bf16* __restrict__ k,
                                                  __bf16* __restrict__ vt) {
  int wave = (blockIdx.x * 256 + threadIdx.x) >> 5;  // 1024*24 = 24576 waves
  int lane = threadIdx.x & 31;
  int half = lane >> 4, lm = lane & 15;
  int tm = wave / 24, tn = wave % 24;                // N-tiles of 64
  const __bf16* arow = A + (size_t)(tm * 16 + lm) * 512;
  const __bf16* brow[4];
  #pragma unroll
  for (int j = 0; j < 4; ++j)
    brow[j] = Bt + (size_t)(tn * 64 + j * 16 + lm) * 512;
  v8f acc[4];
  #pragma unroll
  for (int j = 0; j < 4; ++j) acc[j] = (v8f){0.f,0.f,0.f,0.f,0.f,0.f,0.f,0.f};
  for (int kb = 0; kb < 512; kb += 32) {
    __builtin_prefetch(arow + kb + 128, 0, 1);
    v16bf a = cat8(ldg8(arow + kb + half * 16), ldg8(arow + kb + half * 16 + 8));
    #pragma unroll
    for (int j = 0; j < 4; ++j) {
      v16bf bf = cat8(ldg8(brow[j] + kb + half * 16), ldg8(brow[j] + kb + half * 16 + 8));
      acc[j] = wmma_bf16(a, bf, acc[j]);
    }
  }
  // whole 64-col tile lies inside one of q/k/v and inside one head
  int col0 = tn * 64;
  int g = col0 >> 9, cc0 = col0 & 511, head = cc0 >> 6;
  #pragma unroll
  for (int j = 0; j < 4; ++j) {
    int d = j * 16 + lm;                // 64-col tile == one head's d range
    #pragma unroll
    for (int r = 0; r < 8; ++r) {
      int row = tm * 16 + r + half * 8;
      int b = row >> 10, n = row & 1023;
      __bf16 val = (__bf16)acc[j][r];
      if (g == 0)      q [((size_t)(b * 8 + head) * 1024 + n) * 64 + d] = val;
      else if (g == 1) k [((size_t)(b * 8 + head) * 1024 + n) * 64 + d] = val;
      else             vt[((size_t)(b * 8 + head) * 64 + d) * 1024 + n] = val;
    }
  }
}

// flash attention: one 16-row Q tile per wave, online softmax over 1024 keys
__global__ void __launch_bounds__(256) k_attn(const __bf16* __restrict__ q,
                                              const __bf16* __restrict__ k,
                                              const __bf16* __restrict__ vt,
                                              __bf16* __restrict__ outp) {
  __shared__ __attribute__((aligned(16))) __bf16 pb[8][16 * 32];
  int w    = threadIdx.x >> 5;
  int lane = threadIdx.x & 31;
  int half = lane >> 4, lm = lane & 15;
  int wave = blockIdx.x * 8 + w;        // 8192 waves
  int qt = wave & 63;
  int bh = wave >> 6;                   // b*8+h
  const __bf16* Q = q  + (size_t)bh * 1024 * 64;
  const __bf16* K = k  + (size_t)bh * 1024 * 64;
  const __bf16* V = vt + (size_t)bh * 64 * 1024;

  const __bf16* qrow = Q + (size_t)(qt * 16 + lm) * 64;
  v16bf qf0 = cat8(ldg8(qrow +      half * 16), ldg8(qrow +      half * 16 + 8));
  v16bf qf1 = cat8(ldg8(qrow + 32 + half * 16), ldg8(qrow + 32 + half * 16 + 8));

  v8f o[4];
  float mrun[8], lrun[8];
  #pragma unroll
  for (int dt = 0; dt < 4; ++dt) o[dt] = (v8f){0.f,0.f,0.f,0.f,0.f,0.f,0.f,0.f};
  #pragma unroll
  for (int r = 0; r < 8; ++r) { mrun[r] = -1e30f; lrun[r] = 0.f; }

  for (int kb = 0; kb < 1024; kb += 32) {
    v8f s0 = {0.f,0.f,0.f,0.f,0.f,0.f,0.f,0.f};
    v8f s1 = {0.f,0.f,0.f,0.f,0.f,0.f,0.f,0.f};
    {
      const __bf16* kr0 = K + (size_t)(kb + lm) * 64;
      const __bf16* kr1 = K + (size_t)(kb + 16 + lm) * 64;
      v16bf kf;
      kf = cat8(ldg8(kr0 +      half * 16), ldg8(kr0 +      half * 16 + 8));
      s0 = wmma_bf16(qf0, kf, s0);
      kf = cat8(ldg8(kr0 + 32 + half * 16), ldg8(kr0 + 32 + half * 16 + 8));
      s0 = wmma_bf16(qf1, kf, s0);
      kf = cat8(ldg8(kr1 +      half * 16), ldg8(kr1 +      half * 16 + 8));
      s1 = wmma_bf16(qf0, kf, s1);
      kf = cat8(ldg8(kr1 + 32 + half * 16), ldg8(kr1 + 32 + half * 16 + 8));
      s1 = wmma_bf16(qf1, kf, s1);
    }
    #pragma unroll
    for (int r = 0; r < 8; ++r) {
      float a  = s0[r] * 0.125f;            // 1/sqrt(64)
      float bq = s1[r] * 0.125f;
      float mx = redmax16(fmaxf(a, bq));
      float mnew = fmaxf(mrun[r], mx);
      float sc = __expf(mrun[r] - mnew);
      float p0 = __expf(a - mnew);
      float p1 = __expf(bq - mnew);
      float rs = redsum16(p0 + p1);
      lrun[r] = lrun[r] * sc + rs;
      mrun[r] = mnew;
      #pragma unroll
      for (int dt = 0; dt < 4; ++dt) o[dt][r] *= sc;
      int mrow = r + half * 8;
      pb[w][mrow * 32 + lm]      = (__bf16)p0;
      pb[w][mrow * 32 + 16 + lm] = (__bf16)p1;
    }
    __syncthreads();
    v16bf pf;
    {
      const __bf16* pr = &pb[w][lm * 32];
      pf = cat8(*(const v8bf*)(pr + half * 16), *(const v8bf*)(pr + half * 16 + 8));
    }
    #pragma unroll
    for (int dt = 0; dt < 4; ++dt) {
      const __bf16* vr = V + (size_t)(dt * 16 + lm) * 1024 + kb + half * 16;
      v16bf vf = cat8(ldg8(vr), ldg8(vr + 8));
      o[dt] = wmma_bf16(pf, vf, o[dt]);
    }
    __syncthreads();
  }
  int b = bh >> 3, h = bh & 7;
  #pragma unroll
  for (int r = 0; r < 8; ++r) {
    float inv = 1.0f / lrun[r];
    int n = qt * 16 + r + half * 8;
    size_t orow = ((size_t)b * 1024 + n) * 512 + h * 64;
    #pragma unroll
    for (int dt = 0; dt < 4; ++dt)
      outp[orow + dt * 16 + lm] = (__bf16)(o[dt][r] * inv);
  }
}

// out = attn(16384x512) @ out_w(512x512) + out_b + residual x, write NCHW fp32
// Wave tile: 16 rows x 64 cols.
__global__ void __launch_bounds__(256) k_out_gemm(const __bf16* __restrict__ A,
                                                  const __bf16* __restrict__ Bt,
                                                  const float* __restrict__ bias,
                                                  const float* __restrict__ x,
                                                  float* __restrict__ outp) {
  int wave = (blockIdx.x * 256 + threadIdx.x) >> 5;  // 1024*8 = 8192 waves
  int lane = threadIdx.x & 31;
  int half = lane >> 4, lm = lane & 15;
  int tm = wave >> 3, tn = wave & 7;                 // N-tiles of 64
  const __bf16* arow = A + (size_t)(tm * 16 + lm) * 512;
  const __bf16* brow[4];
  #pragma unroll
  for (int j = 0; j < 4; ++j)
    brow[j] = Bt + (size_t)(tn * 64 + j * 16 + lm) * 512;
  v8f acc[4];
  #pragma unroll
  for (int j = 0; j < 4; ++j) acc[j] = (v8f){0.f,0.f,0.f,0.f,0.f,0.f,0.f,0.f};
  for (int kb = 0; kb < 512; kb += 32) {
    __builtin_prefetch(arow + kb + 128, 0, 1);
    v16bf a = cat8(ldg8(arow + kb + half * 16), ldg8(arow + kb + half * 16 + 8));
    #pragma unroll
    for (int j = 0; j < 4; ++j) {
      v16bf bf = cat8(ldg8(brow[j] + kb + half * 16), ldg8(brow[j] + kb + half * 16 + 8));
      acc[j] = wmma_bf16(a, bf, acc[j]);
    }
  }
  #pragma unroll
  for (int j = 0; j < 4; ++j) {
    int col = tn * 64 + j * 16 + lm;
    float bb = bias[col];
    #pragma unroll
    for (int r = 0; r < 8; ++r) {
      int row = tm * 16 + r + half * 8;
      int b = row >> 10, n = row & 1023;
      size_t xi = ((size_t)(b * 512) + col) * 1024 + n;
      outp[xi] = x[xi] + acc[j][r] + bb;
    }
  }
}

// ---------------- launcher ----------------

extern "C" void kernel_launch(void* const* d_in, const int* in_sizes, int n_in,
                              void* d_out, int out_size, void* d_ws, size_t ws_size,
                              hipStream_t stream) {
  (void)in_sizes; (void)n_in; (void)out_size; (void)ws_size;
  const float* x    = (const float*)d_in[0];
  const float* te   = (const float*)d_in[1];
  const float* w1   = (const float*)d_in[2];
  const float* b1   = (const float*)d_in[3];
  const float* w2   = (const float*)d_in[4];
  const float* b2   = (const float*)d_in[5];
  const float* qkvw = (const float*)d_in[6];
  const float* outw = (const float*)d_in[7];
  const float* outb = (const float*)d_in[8];
  float* out = (float*)d_out;

  char* ws = (char*)d_ws;
  size_t off = 0;
  auto alloc = [&](size_t bytes) -> void* {
    void* p = ws + off;
    off = (off + bytes + 255) & ~(size_t)255;
    return p;
  };
  __bf16* tok    = (__bf16*)alloc((size_t)ROWS * 512 * 2);     // also reused as attn-out
  __bf16* qb     = (__bf16*)alloc((size_t)ROWS * 512 * 2);
  __bf16* kb     = (__bf16*)alloc((size_t)ROWS * 512 * 2);
  __bf16* vtb    = (__bf16*)alloc((size_t)ROWS * 512 * 2);
  __bf16* qkvwt  = (__bf16*)alloc((size_t)1536 * 512 * 2);
  __bf16* outwt  = (__bf16*)alloc((size_t)512 * 512 * 2);
  float*  cosT   = (float*)alloc((size_t)SEQ * HDIM * 4);
  float*  sinT   = (float*)alloc((size_t)SEQ * HDIM * 4);
  float*  stats  = (float*)alloc((size_t)ROWS * 2 * 4);
  float*  gb     = (float*)alloc((size_t)BATCH * 1024 * 4);
  float*  hid    = (float*)alloc((size_t)BATCH * 512 * 4);

  // time-conditioning MLP
  k_mlp1<<<(BATCH * 512) / 256, 256, 0, stream>>>(te, w1, b1, hid);
  k_mlp2<<<(BATCH * 1024) / 256, 256, 0, stream>>>(hid, w2, b2, gb);
  // layernorm stats + AdaCLN tokens (bf16)
  k_stats<<<ROWS / 256, 256, 0, stream>>>(x, stats);
  k_tokens<<<ROWS, 128, 0, stream>>>(x, stats, gb, tok);
  // weight packing (transposed bf16 for B-fragments)
  k_packT<<<(512 * 1536) / 256, 256, 0, stream>>>(qkvw, qkvwt, 512, 1536);
  k_packT<<<(512 * 512) / 256, 256, 0, stream>>>(outw, outwt, 512, 512);
  // RoPE table
  k_ropetab<<<(SEQ * HDIM) / 256, 256, 0, stream>>>(cosT, sinT);
  // QKV projection (WMMA, 16x64 wave tiles): (16384/16)*(1536/64) waves / 8 per block
  k_qkv_gemm<<<(ROWS / 16) * 24 / 8, 256, 0, stream>>>(tok, qkvwt, qb, kb, vtb);
  // RoPE on q,k
  k_rope<<<(BATCH * NHEAD * SEQ * 32) / 256, 256, 0, stream>>>(qb, kb, cosT, sinT);
  // flash attention (WMMA), writes into tok buffer (tokens no longer needed)
  k_attn<<<(BATCH * NHEAD * (SEQ / 16)) / 8, 256, 0, stream>>>(qb, kb, vtb, tok);
  // output projection + bias + residual (WMMA, 16x64 wave tiles)
  k_out_gemm<<<(ROWS / 16) * 8 / 8, 256, 0, stream>>>(tok, outwt, outb, x, out);
}
